// AttentionBlock_2972117369415
// MI455X (gfx1250) — matrix operand order repaired and checked by reading, not compiled, not run on hardware
//
#include <hip/hip_runtime.h>
#include <hip/hip_bf16.h>

typedef __attribute__((ext_vector_type(16))) _Float16 v16h;
typedef __attribute__((ext_vector_type(8)))  _Float16 v8h;
typedef __attribute__((ext_vector_type(2)))  _Float16 half2v;
typedef __attribute__((ext_vector_type(8)))  float    v8f;

#define NB    16      // voxels per block
#define KNN   48      // neighbors
#define CH    64      // channels
#define HEADS 4
#define HD    16
#define FFD   128
#define NW    8       // waves per block
#define NTHR  256

// ---------------- LDS layout (dynamic) ----------------
// half (f16) region offsets, in halves
#define OFF_WQ   0
#define OFF_WK   4096
#define OFF_WV   8192
#define OFF_WO   12288
#define OFF_W1   16384                 // (128,64)
#define OFF_W2   24576                 // (64,128)
#define OFF_X16  32768                 // feats block f16 (16x64)
#define OFF_O16  33792                 // attention out f16 (16x64)
#define OFF_H16  34816                 // ffn hidden f16 (16x128)
#define OFF_XH   36864                 // post-LN1 x f16 (16x64)
#define OFF_KF   37888                 // per-wave kf:  8 * 48*64
#define OFF_KM   (OFF_KF + 24576)      // per-wave K:   8 * 48*64
#define OFF_VM   (OFF_KF + 49152)      // per-wave V:   8 * 48*64
#define HALVES_TOTAL (OFF_KF + 73728)  // 111616 halves
#define HBYTES   (HALVES_TOTAL * 2)    // 223232 B

// float region offsets, in floats (region starts at smem + HBYTES)
#define F_Q    0                       // q*0.25 (16x64)
#define F_Y    1024                    // f32 staging (16x64)
#define F_X    2048                    // post-LN1 x f32 (16x64)
#define F_ATT  3072                    // attn weights per wave: 8 * 4*48
#define F_BK   4608
#define F_BV   4672
#define F_BQ   4736
#define F_BO   4800
#define F_B1   4864                    // 128
#define F_B2   4992
#define F_G1   5056
#define F_BB1  5120
#define F_G2   5184
#define F_BB2  5248
#define FLOATS_TOTAL 5312
#define SMEM_BYTES (HBYTES + FLOATS_TOTAL * 4)   // 244480 B  (< 320 KB WGP LDS)

// ---------------- WMMA helpers ----------------
__device__ __forceinline__ v8f wmma_f16(v16h a, v16h b, v8f c) {
  return __builtin_amdgcn_wmma_f32_16x16x32_f16(false, a, false, b, (short)0, c,
                                                false, false);
}

// A fragment 16x32 f16 from row-major LDS [rows][ldm] at (row0,col0).
// lane L: M=L&15, hi=L>>4; halves K = hi*8+0..7 and 16+hi*8+0..7.
__device__ __forceinline__ v16h load_A(const _Float16* base, int row0, int col0,
                                       int ldm, int lane) {
  int m = lane & 15, hi = lane >> 4;
  const _Float16* p = base + (row0 + m) * ldm + col0 + hi * 8;
  v8h lo = *(const v8h*)p;
  v8h hh = *(const v8h*)(p + 16);
  v16h a;
#pragma unroll
  for (int i = 0; i < 8; ++i) { a[i] = lo[i]; a[i + 8] = hh[i]; }
  return a;
}

// B fragment 32x16: element (k,n) = W[n0+n][k0+k], W row-major [*, ldw].
// lane L: N=L&15, hi=L>>4; K = hi*8+0..7 and 16+hi*8+0..7.
__device__ __forceinline__ v16h load_B(const _Float16* W, int n0, int k0,
                                       int ldw, int lane) {
  int n = lane & 15, hi = lane >> 4;
  const _Float16* p = W + (n0 + n) * ldw + k0 + hi * 8;
  v8h lo = *(const v8h*)p;
  v8h hh = *(const v8h*)(p + 16);
  v16h b;
#pragma unroll
  for (int i = 0; i < 8; ++i) { b[i] = lo[i]; b[i + 8] = hh[i]; }
  return b;
}

// D = A(48x64) @ W^T(64x64) + bias -> dst f16 (48x64)
__device__ __forceinline__ void proj48(const _Float16* src, const _Float16* Wt,
                                       const float* bias, _Float16* dst, int lane) {
  int cl = lane & 15, hi = lane >> 4;
  for (int mt = 0; mt < 3; ++mt) {
    for (int nt = 0; nt < 4; ++nt) {
      v8f acc = {};
#pragma unroll
      for (int kt = 0; kt < 2; ++kt) {
        v16h a = load_A(src, mt * 16, kt * 32, CH, lane);
        v16h b = load_B(Wt, nt * 16, kt * 32, CH, lane);
        acc = wmma_f16(a, b, acc);
      }
      int col = nt * 16 + cl;
      float bv = bias[col];
      int rbase = mt * 16 + hi * 8;
#pragma unroll
      for (int i = 0; i < 8; ++i)
        dst[(rbase + i) * CH + col] = (_Float16)(acc[i] + bv);
    }
  }
}

__device__ __forceinline__ float wave_reduce_add(float v) {
#pragma unroll
  for (int m = 16; m >= 1; m >>= 1) v += __shfl_xor(v, m, 32);
  return v;
}
__device__ __forceinline__ float head_reduce_max(float v) {
  v = fmaxf(v, __shfl_xor(v, 4, 32));
  v = fmaxf(v, __shfl_xor(v, 8, 32));
  v = fmaxf(v, __shfl_xor(v, 16, 32));
  return v;
}
__device__ __forceinline__ float head_reduce_add(float v) {
  v += __shfl_xor(v, 4, 32);
  v += __shfl_xor(v, 8, 32);
  v += __shfl_xor(v, 16, 32);
  return v;
}

__global__ __launch_bounds__(NTHR)
void attn_block_kernel(const float* __restrict__ feats,
                       const float* __restrict__ coords,
                       const int* __restrict__ key_indices,
                       const unsigned char* __restrict__ key_mask,
                       const float* __restrict__ w_q, const float* __restrict__ b_q,
                       const float* __restrict__ w_k, const float* __restrict__ b_k,
                       const float* __restrict__ w_v, const float* __restrict__ b_v,
                       const float* __restrict__ w_o, const float* __restrict__ b_o,
                       const float* __restrict__ pos_w, const float* __restrict__ pos_b,
                       const float* __restrict__ w1, const float* __restrict__ b1,
                       const float* __restrict__ w2, const float* __restrict__ b2,
                       const float* __restrict__ ln1_g, const float* __restrict__ ln1_b,
                       const float* __restrict__ ln2_g, const float* __restrict__ ln2_b,
                       float* __restrict__ out, int N) {
  extern __shared__ __align__(16) char smem_raw[];
  _Float16* smh = (_Float16*)smem_raw;
  float*    smf = (float*)(smem_raw + HBYTES);

  const int tid  = threadIdx.x;
  const int lane = tid & 31;
  const int w    = tid >> 5;
  const int base = blockIdx.x * NB;

  // ---- stage weights (f32 -> f16) and biases into LDS ----
  for (int i = tid; i < CH * CH; i += NTHR) {
    smh[OFF_WQ + i] = (_Float16)w_q[i];
    smh[OFF_WK + i] = (_Float16)w_k[i];
    smh[OFF_WV + i] = (_Float16)w_v[i];
    smh[OFF_WO + i] = (_Float16)w_o[i];
  }
  for (int i = tid; i < FFD * CH; i += NTHR) {
    smh[OFF_W1 + i] = (_Float16)w1[i];
    smh[OFF_W2 + i] = (_Float16)w2[i];
  }
  for (int i = tid; i < NB * CH; i += NTHR) {
    int n = base + i / CH;
    smh[OFF_X16 + i] = (n < N) ? (_Float16)feats[(size_t)n * CH + i % CH]
                               : (_Float16)0.f;
  }
  if (tid < CH) {
    smf[F_BK + tid] = b_k[tid];  smf[F_BV + tid] = b_v[tid];
    smf[F_BQ + tid] = b_q[tid];  smf[F_BO + tid] = b_o[tid];
    smf[F_B2 + tid] = b2[tid];
    smf[F_G1 + tid] = ln1_g[tid]; smf[F_BB1 + tid] = ln1_b[tid];
    smf[F_G2 + tid] = ln2_g[tid]; smf[F_BB2 + tid] = ln2_b[tid];
  }
  if (tid < FFD) smf[F_B1 + tid] = b1[tid];
  __syncthreads();

  const int cl = lane & 15, hi = lane >> 4, rbase16 = hi * 8;

  // ---- wave 0: batched Q = X @ Wq^T, pre-scaled by HD^-0.5 ----
  if (w == 0) {
    for (int nt = 0; nt < 4; ++nt) {
      v8f acc = {};
#pragma unroll
      for (int kt = 0; kt < 2; ++kt) {
        v16h a = load_A(smh + OFF_X16, 0, kt * 32, CH, lane);
        v16h b = load_B(smh + OFF_WQ, nt * 16, kt * 32, CH, lane);
        acc = wmma_f16(a, b, acc);
      }
      int col = nt * 16 + cl;
      float bq = smf[F_BQ + col];
#pragma unroll
      for (int i = 0; i < 8; ++i)
        smf[F_Q + (rbase16 + i) * CH + col] = (acc[i] + bq) * 0.25f;
    }
  }

  // ---- per-wave: gather + pos-enc + K/V projections (2 voxels/wave) ----
  _Float16* kfw = smh + OFF_KF + w * (KNN * CH);
  _Float16* kmw = smh + OFF_KM + w * (KNN * CH);
  _Float16* vmw = smh + OFF_VM + w * (KNN * CH);

  const int c0 = lane * 2, c1 = c0 + 1;
  const float pw00 = pos_w[c0 * 3], pw01 = pos_w[c0 * 3 + 1], pw02 = pos_w[c0 * 3 + 2];
  const float pw10 = pos_w[c1 * 3], pw11 = pos_w[c1 * 3 + 1], pw12 = pos_w[c1 * 3 + 2];
  const float pb0 = pos_b[c0], pb1 = pos_b[c1];

  for (int sub = 0; sub < 2; ++sub) {
    const int vi = w * 2 + sub;
    const int n  = base + vi;
    if (n >= N) continue;

    // prefetch gathered feature rows (global_prefetch_b8)
    for (int r = lane; r < KNN; r += 32) {
      int idx = key_indices[(size_t)n * KNN + r];
      __builtin_prefetch(feats + (size_t)idx * CH, 0, 0);
    }

    const float qx = coords[(size_t)n * 3], qy = coords[(size_t)n * 3 + 1],
                qz = coords[(size_t)n * 3 + 2];
    for (int r = 0; r < KNN; ++r) {
      int idx = key_indices[(size_t)n * KNN + r];
      float rx = coords[(size_t)idx * 3]     - qx;
      float ry = coords[(size_t)idx * 3 + 1] - qy;
      float rz = coords[(size_t)idx * 3 + 2] - qz;
      const float2* f2 = (const float2*)(feats + (size_t)idx * CH);
      float2 fv = f2[lane];
      float p0 = fmaxf(0.f, pw00 * rx + pw01 * ry + pw02 * rz + pb0);
      float p1 = fmaxf(0.f, pw10 * rx + pw11 * ry + pw12 * rz + pb1);
      half2v hv; hv[0] = (_Float16)(fv.x + p0); hv[1] = (_Float16)(fv.y + p1);
      ((half2v*)(kfw + r * CH))[lane] = hv;
    }
    // K = kf @ Wk^T + bk ; V = kf @ Wv^T + bv   (24 WMMAs each)
    proj48(kfw, smh + OFF_WK, smf + F_BK, kmw, lane);
    proj48(kfw, smh + OFF_WV, smf + F_BV, vmw, lane);
  }
  __syncthreads();   // Qf visible to all waves

  // ---- per-wave: scores -> softmax -> attn-weighted V ----
  for (int sub = 0; sub < 2; ++sub) {
    const int vi = w * 2 + sub;
    const int n  = base + vi;
    if (n >= N) continue;

    const int h = lane & 3;       // head
    const int g = lane >> 2;      // 8 lanes per head cover 48 keys
    float qv[HD];
#pragma unroll
    for (int d = 0; d < HD; ++d) qv[d] = smf[F_Q + vi * CH + h * HD + d];

    float sc[6];
#pragma unroll
    for (int j = 0; j < 6; ++j) {
      int kk = g + 8 * j;
      const _Float16* kp = kmw + kk * CH + h * HD;
      v8h k0 = *(const v8h*)kp;
      v8h k1 = *(const v8h*)(kp + 8);
      float s = 0.f;
#pragma unroll
      for (int d = 0; d < 8; ++d) s += qv[d] * (float)k0[d] + qv[8 + d] * (float)k1[d];
      if (key_mask[(size_t)n * KNN + kk]) s = -1e9f;
      sc[j] = s;
    }
    float mx = sc[0];
#pragma unroll
    for (int j = 1; j < 6; ++j) mx = fmaxf(mx, sc[j]);
    mx = head_reduce_max(mx);
    float lsum = 0.f;
#pragma unroll
    for (int j = 0; j < 6; ++j) { sc[j] = __expf(sc[j] - mx); lsum += sc[j]; }
    float inv = 1.f / head_reduce_add(lsum);
#pragma unroll
    for (int j = 0; j < 6; ++j)
      smf[F_ATT + w * (HEADS * KNN) + h * KNN + (g + 8 * j)] = sc[j] * inv;

    asm volatile("s_wait_dscnt 0" ::: "memory");  // wave-internal LDS handoff

    const int hh = lane >> 3;     // head owning cols c0,c1
    float a0 = 0.f, a1 = 0.f;
    for (int kk = 0; kk < KNN; ++kk) {
      float at = smf[F_ATT + w * (HEADS * KNN) + hh * KNN + kk];
      half2v pv = *(const half2v*)(vmw + kk * CH + c0);
      a0 += at * (float)pv[0];
      a1 += at * (float)pv[1];
    }
    smh[OFF_O16 + vi * CH + c0] = (_Float16)a0;
    smh[OFF_O16 + vi * CH + c1] = (_Float16)a1;
  }
  __syncthreads();

  // ---- wave 0: output projection + bias + residual -> fY ----
  if (w == 0) {
    for (int nt = 0; nt < 4; ++nt) {
      v8f acc = {};
#pragma unroll
      for (int kt = 0; kt < 2; ++kt) {
        v16h a = load_A(smh + OFF_O16, 0, kt * 32, CH, lane);
        v16h b = load_B(smh + OFF_WO, nt * 16, kt * 32, CH, lane);
        acc = wmma_f16(a, b, acc);
      }
      int col = nt * 16 + cl;
      float bo = smf[F_BO + col];
#pragma unroll
      for (int i = 0; i < 8; ++i) {
        int row = rbase16 + i;
        int n = base + row;
        float res = (n < N) ? feats[(size_t)n * CH + col] : 0.f;
        smf[F_Y + row * CH + col] = acc[i] + bo + res;
      }
    }
  }
  __syncthreads();

  // ---- LN1 (2 rows per wave) -> fX (f32) and XH (f16) ----
#pragma unroll
  for (int rr = 0; rr < 2; ++rr) {
    int r = w * 2 + rr;
    float x0 = smf[F_Y + r * CH + c0], x1 = smf[F_Y + r * CH + c1];
    float s  = wave_reduce_add(x0 + x1);
    float sq = wave_reduce_add(x0 * x0 + x1 * x1);
    float mean = s * (1.f / CH);
    float var  = sq * (1.f / CH) - mean * mean;
    float rstd = rsqrtf(var + 1e-5f);
    float y0 = (x0 - mean) * rstd * smf[F_G1 + c0] + smf[F_BB1 + c0];
    float y1 = (x1 - mean) * rstd * smf[F_G1 + c1] + smf[F_BB1 + c1];
    smf[F_X + r * CH + c0] = y0;            smf[F_X + r * CH + c1] = y1;
    smh[OFF_XH + r * CH + c0] = (_Float16)y0;
    smh[OFF_XH + r * CH + c1] = (_Float16)y1;
  }
  __syncthreads();

  // ---- FFN1: h = relu(x @ W1^T + b1), n-tile = wave id ----
  {
    v8f acc = {};
#pragma unroll
    for (int kt = 0; kt < 2; ++kt) {
      v16h a = load_A(smh + OFF_XH, 0, kt * 32, CH, lane);
      v16h b = load_B(smh + OFF_W1, w * 16, kt * 32, CH, lane);
      acc = wmma_f16(a, b, acc);
    }
    int col = w * 16 + cl;
    float bb = smf[F_B1 + col];
#pragma unroll
    for (int i = 0; i < 8; ++i)
      smh[OFF_H16 + (rbase16 + i) * FFD + col] =
          (_Float16)fmaxf(0.f, acc[i] + bb);
  }
  __syncthreads();

  // ---- FFN2: y = h @ W2^T + b2 + x, n-tiles on waves 0..3 ----
  if (w < 4) {
    v8f acc = {};
#pragma unroll
    for (int kt = 0; kt < 4; ++kt) {
      v16h a = load_A(smh + OFF_H16, 0, kt * 32, FFD, lane);
      v16h b = load_B(smh + OFF_W2, w * 16, kt * 32, FFD, lane);
      acc = wmma_f16(a, b, acc);
    }
    int col = w * 16 + cl;
    float bb = smf[F_B2 + col];
#pragma unroll
    for (int i = 0; i < 8; ++i) {
      int row = rbase16 + i;
      smf[F_Y + row * CH + col] = acc[i] + bb + smf[F_X + row * CH + col];
    }
  }
  __syncthreads();

  // ---- LN2 (2 rows per wave) -> global out ----
#pragma unroll
  for (int rr = 0; rr < 2; ++rr) {
    int r = w * 2 + rr;
    int n = base + r;
    float x0 = smf[F_Y + r * CH + c0], x1 = smf[F_Y + r * CH + c1];
    float s  = wave_reduce_add(x0 + x1);
    float sq = wave_reduce_add(x0 * x0 + x1 * x1);
    float mean = s * (1.f / CH);
    float var  = sq * (1.f / CH) - mean * mean;
    float rstd = rsqrtf(var + 1e-5f);
    if (n < N) {
      out[(size_t)n * CH + c0] =
          (x0 - mean) * rstd * smf[F_G2 + c0] + smf[F_BB2 + c0];
      out[(size_t)n * CH + c1] =
          (x1 - mean) * rstd * smf[F_G2 + c1] + smf[F_BB2 + c1];
    }
  }
}

extern "C" void kernel_launch(void* const* d_in, const int* in_sizes, int n_in,
                              void* d_out, int out_size, void* d_ws, size_t ws_size,
                              hipStream_t stream) {
  (void)n_in; (void)out_size; (void)d_ws; (void)ws_size;
  const float* feats  = (const float*)d_in[0];
  const float* coords = (const float*)d_in[1];
  const int*   kidx   = (const int*)d_in[2];
  const unsigned char* kmask = (const unsigned char*)d_in[3];
  const float* w_q = (const float*)d_in[4];  const float* b_q = (const float*)d_in[5];
  const float* w_k = (const float*)d_in[6];  const float* b_k = (const float*)d_in[7];
  const float* w_v = (const float*)d_in[8];  const float* b_v = (const float*)d_in[9];
  const float* w_o = (const float*)d_in[10]; const float* b_o = (const float*)d_in[11];
  const float* pos_w = (const float*)d_in[12]; const float* pos_b = (const float*)d_in[13];
  const float* w1 = (const float*)d_in[14]; const float* b1 = (const float*)d_in[15];
  const float* w2 = (const float*)d_in[16]; const float* b2 = (const float*)d_in[17];
  const float* g1 = (const float*)d_in[18]; const float* bb1 = (const float*)d_in[19];
  const float* g2 = (const float*)d_in[20]; const float* bb2 = (const float*)d_in[21];

  const int N = in_sizes[0] / CH;
  (void)hipFuncSetAttribute((const void*)attn_block_kernel,
                            hipFuncAttributeMaxDynamicSharedMemorySize, SMEM_BYTES);
  const int blocks = (N + NB - 1) / NB;
  attn_block_kernel<<<blocks, NTHR, SMEM_BYTES, stream>>>(
      feats, coords, kidx, kmask, w_q, b_q, w_k, b_k, w_v, b_v, w_o, b_o,
      pos_w, pos_b, w1, b1, w2, b2, g1, bb1, g2, bb2, (float*)d_out, N);
}
